// CondLaneNetONNX_45140106281354
// MI455X (gfx1250) — matrix operand order repaired and checked by reading, not compiled, not run on hardware
//
#include <hip/hip_runtime.h>
#include <hip/hip_bf16.h>

// ---------------- problem constants ----------------
#define CIN    256
#define HMSK   272
#define WMSK   480
#define HWM    (HMSK * WMSK)      // 130560
#define HHM    136
#define WHM    240
#define HWH    (HHM * WHM)        // 32640
#define MBCH   64
#define NMP    67                 // NUM_MASK_PARAMS
#define NGP    134                // NUM_GEN_PARAMS
#define NI     4                  // num_ins

typedef __attribute__((ext_vector_type(16))) __bf16 v16bf;
typedef __attribute__((ext_vector_type(8)))  __bf16 v8bf;
typedef __attribute__((ext_vector_type(2)))  __bf16 v2bf;
typedef __attribute__((ext_vector_type(8)))  float  v8f;
typedef __attribute__((ext_vector_type(4)))  float  v4f;

// LDS pixel stride (elements): 288*2B = 576B -> 32B-aligned v16bf fragment loads,
// and 144-dword pixel stride keeps staging stores conflict-free.
#define CSTR   288
#define ROWELE (34 * CSTR)        // LDS elements per staged input row

// ---------------- K0: pre-swizzle conv weights into WMMA A-fragment layout ----------------
// Layout: [dydx 0..8][cb 0..7][otile 0..3][lane 0..31][e 0..15] bf16 (32B/lane contiguous)
// A 16x32 bf16 k order used here (same as the B loads): M = lane%16, half = lane/16,
//   K = 16*half + e
__global__ void k_swizzle_w(const float* __restrict__ mb_w, __bf16* __restrict__ wz) {
    int g = blockIdx.x * 256 + threadIdx.x;
    if (g >= 9 * 8 * 4 * 32 * 16) return;
    int e     = g & 15;
    int lane  = (g >> 4) & 31;
    int otile = (g >> 9) & 3;
    int cb    = (g >> 11) & 7;
    int dydx  = g >> 14;
    int M  = lane & 15, hf = lane >> 4;
    int v  = e >> 1;
    int K  = 16 * hf + e;
    int o  = otile * 16 + M;
    int c  = cb * 32 + K;
    int ky = dydx / 3, kx = dydx % 3;
    wz[g] = (__bf16)mb_w[((o * CIN + c) * 3 + ky) * 3 + kx];
}

// ---------------- K1: heatmap GEMV + sigmoid + clip (4 px / thread, b128 loads) ----------
__global__ void k_heat(const float* __restrict__ out1, const float* __restrict__ hm_w,
                       const float* __restrict__ hm_b, float* __restrict__ heat) {
    int q = blockIdx.x * 256 + threadIdx.x;        // group of 4 pixels
    if (q >= HWH / 4) return;
    float b = hm_b[0];
    v4f acc = {b, b, b, b};
    for (int c = 0; c < CIN; ++c) {
        v4f v = *(const v4f*)(out1 + c * HWH + q * 4);
        acc += hm_w[c] * v;
    }
    v4f s;
    #pragma unroll
    for (int j = 0; j < 4; ++j) {
        float t = 1.f / (1.f + __expf(-acc[j]));
        s[j] = fminf(fmaxf(t, 1e-4f), 1.f - 1e-4f);
    }
    *(v4f*)(heat + q * 4) = s;
}

// ---------------- K2: 3x3 NMS + top-4 (single block) ----------------
__global__ void k_nms_topk(const float* __restrict__ heat, float* __restrict__ d_out,
                           int* __restrict__ inds_ws) {
    __shared__ float sv[1024];
    __shared__ int   si[1024];
    int t = threadIdx.x;
    float lv[4] = {-1e30f, -1e30f, -1e30f, -1e30f};
    int   li[4] = {0, 0, 0, 0};
    for (int p = t; p < HWH; p += 256) {
        int py = p / WHM, px = p % WHM;
        float c0 = heat[p];
        float m = c0;
        for (int dy = -1; dy <= 1; ++dy)
            for (int dx = -1; dx <= 1; ++dx) {
                int ny = py + dy, nx = px + dx;
                if (ny >= 0 && ny < HHM && nx >= 0 && nx < WHM)
                    m = fmaxf(m, heat[ny * WHM + nx]);
            }
        float val = (c0 == m) ? c0 : 0.f;
        if (val > lv[3]) {
            int k = 3;
            while (k > 0 && val > lv[k - 1]) { lv[k] = lv[k - 1]; li[k] = li[k - 1]; --k; }
            lv[k] = val; li[k] = p;
        }
    }
    for (int k = 0; k < 4; ++k) { sv[t * 4 + k] = lv[k]; si[t * 4 + k] = li[k]; }
    __syncthreads();
    if (t == 0) {
        for (int k = 0; k < NI; ++k) {
            float best = -1e30f; int bidx = (1 << 30), bj = 0;
            for (int j = 0; j < 1024; ++j) {
                if (sv[j] > best || (sv[j] == best && si[j] < bidx)) {
                    best = sv[j]; bidx = si[j]; bj = j;
                }
            }
            d_out[k]      = best;
            d_out[NI + k] = (float)bidx;
            inds_ws[k]    = bidx;
            sv[bj] = -1e30f;
        }
    }
}

// ---------------- K3: gen_params for the 4 selected pixels only ----------------
__global__ void k_gen_params(const float* __restrict__ out1, const float* __restrict__ pw,
                             const float* __restrict__ pb, const int* __restrict__ inds,
                             float* __restrict__ gp) {
    __shared__ float col[CIN];
    int i = blockIdx.x, t = threadIdx.x;
    int p = inds[i];
    col[t] = out1[t * HWH + p];
    __syncthreads();
    if (t < NGP) {
        float acc = pb[t];
        for (int c = 0; c < CIN; ++c)
            acc += pw[t * CIN + c] * col[c];
        gp[i * NGP + t] = acc;
    }
}

// ---------------- K4: conv3x3 + ReLU via bf16 WMMA -------------------------------------
// Block: 256 threads = 8 waves, tile = 64 outch x 32 px x 2 output rows.
// Stage 4 input rows once (single barrier); each wave: 2 accumulators (one per out row),
// A-fragment loaded once per (tap,cb) and reused for both rows -> 144 WMMAs / wave.
// grid = (480/32, 272/2).
__global__ __launch_bounds__(256, 2)
void k_conv_wmma(const float* __restrict__ out0, const __bf16* __restrict__ wz,
                 const float* __restrict__ mb_b, __bf16* __restrict__ mbuf) {
    __shared__ __bf16 sx[4 * ROWELE];               // 4 rows x [px 0..33][c 0..255], padded
    const int x0 = blockIdx.x * 32;
    const int y0 = blockIdx.y * 2;
    const int t  = threadIdx.x;
    const int lane  = t & 31;
    const int wave  = t >> 5;
    const int otile = wave & 3;
    const int ptile = wave >> 2;
    const int nn = lane & 15;
    const int hf = lane >> 4;

    // ---- stage 4 input rows (y0-1 .. y0+2) as bf16, transposed [px][c] ----
    for (int rr = 0; rr < 4; ++rr) {
        const int ysrc = y0 - 1 + rr;
        const bool ok  = (ysrc >= 0) && (ysrc < HMSK);
        __bf16* dst = sx + rr * ROWELE;
        // core 32 px: channel-paired float4 loads -> packed v2bf dword stores
        for (int k = 0; k < 4; ++k) {
            int pid = k * 256 + t;                   // 0..1023
            int c2  = pid >> 3;                      // channel pair 0..127
            int xv  = (pid & 7) * 4;                 // 0,4,...,28
            v4f lo = {0.f, 0.f, 0.f, 0.f}, hi = lo;
            if (ok) {
                lo = *(const v4f*)(out0 + (2 * c2 + 0) * HWM + ysrc * WMSK + x0 + xv);
                hi = *(const v4f*)(out0 + (2 * c2 + 1) * HWM + ysrc * WMSK + x0 + xv);
            }
            #pragma unroll
            for (int j = 0; j < 4; ++j) {
                v2bf pk = {(__bf16)lo[j], (__bf16)hi[j]};
                *(v2bf*)(dst + (xv + j + 1) * CSTR + 2 * c2) = pk;
            }
        }
        // halo columns (px 0 and 33), one channel per thread
        float vl = (ok && x0 > 0)         ? out0[t * HWM + ysrc * WMSK + x0 - 1]  : 0.f;
        float vr = (ok && x0 + 32 < WMSK) ? out0[t * HWM + ysrc * WMSK + x0 + 32] : 0.f;
        dst[0 * CSTR + t]  = (__bf16)vl;
        dst[33 * CSTR + t] = (__bf16)vr;
    }
    // speculative prefetch of input row y0+3 (first row past this block's staged rows)
    {
        int ynext = y0 + 3;
        if (ynext < HMSK)
            __builtin_prefetch(&out0[t * HWM + ynext * WMSK + x0], 0, 1);
    }
    __syncthreads();

    // ---- 144 WMMAs per wave: A reused across the two output rows ----
    const v16bf* wzf = (const v16bf*)wz;
    v8f acc0 = {0.f, 0.f, 0.f, 0.f, 0.f, 0.f, 0.f, 0.f};
    v8f acc1 = acc0;
    for (int dy3 = 0; dy3 < 3; ++dy3) {
        for (int dx = 0; dx < 3; ++dx) {
            const int px    = ptile * 16 + nn + dx;   // 0..33
            const int dydx  = dy3 * 3 + dx;
            const __bf16* b0 = sx + (dy3 + 0) * ROWELE + px * CSTR + hf * 16;
            const __bf16* b1 = sx + (dy3 + 1) * ROWELE + px * CSTR + hf * 16;
            #pragma unroll
            for (int cb = 0; cb < 8; ++cb) {
                v16bf a  = wzf[((dydx * 8 + cb) * 4 + otile) * 32 + lane];
                v16bf f0 = *(const v16bf*)(b0 + cb * 32);
                v16bf f1 = *(const v16bf*)(b1 + cb * 32);
                acc0 = __builtin_amdgcn_wmma_f32_16x16x32_bf16(
                           false, a, false, f0, (short)0, acc0, false, false);
                acc1 = __builtin_amdgcn_wmma_f32_16x16x32_bf16(
                           false, a, false, f1, (short)0, acc1, false, false);
            }
        }
    }

    // ---- bias + relu + packed store: mbuf layout [pixel][64ch] bf16 ----
    // C/D layout: VGPR v -> M = v + 8*hf, N = lane%16  => lane owns 8 consecutive channels
    const int o0 = otile * 16;
    const int xo = x0 + ptile * 16 + nn;
    const float* bb = mb_b + o0 + hf * 8;
    {
        v8bf ov;
        #pragma unroll
        for (int v = 0; v < 8; ++v)
            ov[v] = (__bf16)fmaxf(acc0[v] + bb[v], 0.f);
        *(v8bf*)(mbuf + (size_t)((y0 + 0) * WMSK + xo) * MBCH + o0 + hf * 8) = ov;
    }
    {
        v8bf ov;
        #pragma unroll
        for (int v = 0; v < 8; ++v)
            ov[v] = (__bf16)fmaxf(acc1[v] + bb[v], 0.f);
        *(v8bf*)(mbuf + (size_t)((y0 + 1) * WMSK + xo) * MBCH + o0 + hf * 8) = ov;
    }
}

// ---------------- K5: dynamic heads (masks0 + regs0), [px][ch] bf16 input ----------------
__global__ void k_dyn_head(const __bf16* __restrict__ mbuf, const float* __restrict__ gp_g,
                           float* __restrict__ d_out) {
    __shared__ float gp[NI * NGP];
    int t = threadIdx.x;
    for (int j = t; j < NI * NGP; j += 256) gp[j] = gp_g[j];
    __syncthreads();
    int p = blockIdx.x * 256 + t;
    if (p >= HWM) return;
    int py = p / WMSK, px = p % WMSK;
    float cx = -1.f + 2.f * px / (float)(WMSK - 1);
    float cy = -1.f + 2.f * py / (float)(HMSK - 1);
    float am[NI], ar[NI];
    #pragma unroll
    for (int i = 0; i < NI; ++i) { am[i] = gp[i * NGP + 66]; ar[i] = gp[i * NGP + 133]; }
    const __bf16* row = mbuf + (size_t)p * MBCH;
    #pragma unroll
    for (int cb = 0; cb < 8; ++cb) {
        v8bf vv = *(const v8bf*)(row + cb * 8);
        #pragma unroll
        for (int j = 0; j < 8; ++j) {
            float v = (float)vv[j];
            int c = cb * 8 + j;
            #pragma unroll
            for (int i = 0; i < NI; ++i) {
                am[i] += gp[i * NGP + c] * v;
                ar[i] += gp[i * NGP + NMP + c] * v;
            }
        }
    }
    #pragma unroll
    for (int i = 0; i < NI; ++i) {
        am[i] += gp[i * NGP + 64] * cx + gp[i * NGP + 65] * cy;
        ar[i] += gp[i * NGP + NMP + 64] * cx + gp[i * NGP + NMP + 65] * cy;
        d_out[2 * NI + i * HWM + p] = ar[i];                 // regs0
        d_out[2 * NI + NI * HWM + i * HWM + p] = am[i];      // masks0
    }
}

// ---------------- K6: per-column MLP on masks0 (feat_range) ----------------
__global__ void k_mlp(const float* __restrict__ masks, const float* __restrict__ w1,
                      const float* __restrict__ b1, const float* __restrict__ w2,
                      const float* __restrict__ b2, float* __restrict__ outf) {
    int g = blockIdx.x * 128 + threadIdx.x;
    if (g >= NI * WMSK) return;
    int i = g / WMSK, x = g % WMSK;
    const float* m = masks + i * HWM + x;
    float h[64];
    #pragma unroll
    for (int j = 0; j < 64; ++j) h[j] = b1[j];
    for (int yv = 0; yv < HMSK; ++yv) {
        float v = m[yv * WMSK];
        const float* wr = w1 + yv * 64;
        #pragma unroll
        for (int j = 0; j < 64; ++j) h[j] += v * wr[j];
    }
    float o0 = b2[0], o1 = b2[1];
    #pragma unroll
    for (int j = 0; j < 64; ++j) {
        float hr = fmaxf(h[j], 0.f);
        o0 += hr * w2[j * 2 + 0];
        o1 += hr * w2[j * 2 + 1];
    }
    outf[g * 2 + 0] = o0;
    outf[g * 2 + 1] = o1;
}

// ---------------- launcher ----------------
extern "C" void kernel_launch(void* const* d_in, const int* in_sizes, int n_in,
                              void* d_out, int out_size, void* d_ws, size_t ws_size,
                              hipStream_t stream) {
    const float* out0   = (const float*)d_in[0];
    const float* out1   = (const float*)d_in[1];
    const float* hm_w   = (const float*)d_in[2];
    const float* hm_b   = (const float*)d_in[3];
    const float* pw     = (const float*)d_in[4];
    const float* pb     = (const float*)d_in[5];
    const float* mb_w   = (const float*)d_in[6];
    const float* mb_b   = (const float*)d_in[7];
    const float* mlp_w1 = (const float*)d_in[8];
    const float* mlp_b1 = (const float*)d_in[9];
    const float* mlp_w2 = (const float*)d_in[10];
    const float* mlp_b2 = (const float*)d_in[11];
    float* out = (float*)d_out;

    // workspace layout (bytes)
    char* ws = (char*)d_ws;
    float*  heat   = (float*)(ws + 0);                       // 32640 f32
    int*    inds   = (int*)(ws + 130560);                    // 4 i32
    float*  gp     = (float*)(ws + 130576);                  // 4*134 f32
    __bf16* wz     = (__bf16*)(ws + 133120);                 // 147456 bf16 (swizzled W)
    __bf16* mbuf   = (__bf16*)(ws + 428032);                 // 130560*64 bf16, [px][ch]

    // K0: weight swizzle (147456 elements)
    k_swizzle_w<<<576, 256, 0, stream>>>(mb_w, wz);
    // K1: heatmap GEMV + sigmoid + clip (4 px / thread)
    k_heat<<<(HWH / 4 + 255) / 256, 256, 0, stream>>>(out1, hm_w, hm_b, heat);
    // K2: NMS + top-4
    k_nms_topk<<<1, 256, 0, stream>>>(heat, out, inds);
    // K3: gen_params at the 4 selected pixels
    k_gen_params<<<NI, 256, 0, stream>>>(out1, pw, pb, inds, gp);
    // K4: conv3x3 + relu via WMMA bf16 (2 output rows per block)
    k_conv_wmma<<<dim3(WMSK / 32, HMSK / 2), 256, 0, stream>>>(out0, wz, mb_b, mbuf);
    // K5: dynamic heads -> regs0, masks0 in d_out
    k_dyn_head<<<HWM / 256, 256, 0, stream>>>(mbuf, gp, out);
    // K6: MLP on masks0 columns -> feat_range
    const float* masks = out + 2 * NI + NI * HWM;
    float* featr = out + 2 * NI + 2 * NI * HWM;
    k_mlp<<<(NI * WMSK + 127) / 128, 128, 0, stream>>>(masks, mlp_w1, mlp_b1, mlp_w2, mlp_b2, featr);
}
